// SAVSE_59828894433538
// MI455X (gfx1250) — matrix operand order repaired and checked by reading, not compiled, not run on hardware
//
#include <hip/hip_runtime.h>
#include <math.h>

// ---------------------------------------------------------------------------
// CDNA5 (gfx1250) WMMA types
// ---------------------------------------------------------------------------
typedef __attribute__((ext_vector_type(16))) _Float16 v16h;
typedef __attribute__((ext_vector_type(8)))  float    v8f;

#define DEVINL __device__ __forceinline__

// ISA 16-bit A-matrix 16x32 layout: lane = row M (lane&15); lane bit4 selects
// K-half (+8).  VGPR v<4 holds K = 2v,2v+1 ; VGPR v>=4 holds K = 16+2(v-4)...
DEVINL int a_k_of(int lane, int e) {
  int half = lane >> 4;
  int v = e >> 1, lo = e & 1;
  int k = (v < 4) ? (v * 2 + lo) : (16 + (v - 4) * 2 + lo);
  return k + half * 8;
}
// ISA B-matrix 32x16 layout: n = lane&15 ; k = 8*(lane>>4) + e (e<8), 16 + 8*(lane>>4) + (e-8) (e>=8)
DEVINL int b_k_of(int lane, int e) { return ((lane >> 4) * 8) + (e < 8 ? e : e + 8); }
// ISA D-matrix 16x16 f32 layout: n = lane&15 ; m = vgpr + 8*(lane>>4)

// ---------------------------------------------------------------------------
// Generic strided, two-level-batched GEMM:  C = alpha * A @ B (+ bias[n])
//   batch z -> b1 = z/inner, b2 = z%inner ; base += b1*bs?1 + b2*bs?2
// One wave (32 lanes) per 16x16 output tile; blockDim = (32,4).
// Operand fetch is fully branch-free: clamped addresses + mask-MULTIPLY
// (never a select feeding only a conditional use, which clang would sink
// into an exec-masked block with a per-load wait).
// ---------------------------------------------------------------------------
__global__ void wmma_gemm_kernel(
    const float* __restrict__ A, const float* __restrict__ Bm,
    const float* __restrict__ bias, float* __restrict__ C,
    int M, int N, int K,
    int sAm, int sAk, int sBk, int sBn, int sCm, int sCn,
    long bsA1, long bsB1, long bsC1,
    int inner, long bsA2, long bsB2, long bsC2,
    float alpha)
{
  const int lane = threadIdx.x;
  const int m0 = blockIdx.x * 16;
  const int n0 = (blockIdx.y * blockDim.y + threadIdx.y) * 16;
  if (n0 >= N) return;                       // wave-uniform exit

  const int z  = blockIdx.z;
  const int b1 = z / inner, b2 = z - b1 * inner;
  const float* Ab = A  + b1 * bsA1 + b2 * bsA2;
  const float* Bb = Bm + b1 * bsB1 + b2 * bsB2;
  float*       Cb = C  + b1 * bsC1 + b2 * bsC2;

  const int  am  = m0 + (lane & 15);
  const bool amv = (am < M);
  const int  amc = amv ? am : (M - 1);
  const int  bn  = n0 + (lane & 15);
  const bool bnv = (bn < N);
  const int  bnc = bnv ? bn : (N - 1);

  const float maskA = amv ? 1.f : 0.f;       // v_cndmask of constants, no branch
  const float maskB = bnv ? 1.f : 0.f;

  const float* Arow = Ab + (long)amc * sAm;
  const float* Bcol = Bb + (long)bnc * sBn;

  // Full tile: k0+32 <= K guarantees every ka,kb < K (intra-tile offset <= 31).
  auto load_full = [&](int k0, v16h& a, v16h& b) {
#pragma unroll
    for (int e = 0; e < 16; ++e) {
      float xa = Arow[(long)(k0 + a_k_of(lane, e)) * sAk];
      a[e] = (_Float16)(xa * maskA);
      float xb = Bcol[(long)(k0 + b_k_of(lane, e)) * sBk];
      b[e] = (_Float16)(xb * maskB);
    }
  };
  // Tail tile: clamp k and fold the K-bound into the multiply mask.
  auto load_tail = [&](int k0, v16h& a, v16h& b) {
#pragma unroll
    for (int e = 0; e < 16; ++e) {
      int ka = k0 + a_k_of(lane, e);
      float xa = Arow[(long)(ka < K ? ka : K - 1) * sAk];
      a[e] = (_Float16)(xa * ((ka < K) ? maskA : 0.f));
      int kb = k0 + b_k_of(lane, e);
      float xb = Bcol[(long)(kb < K ? kb : K - 1) * sBk];
      b[e] = (_Float16)(xb * ((kb < K) ? maskB : 0.f));
    }
  };

  v8f acc = {};
  int k0 = 0;
  for (; k0 + 64 <= K; k0 += 64) {
    __builtin_prefetch(&Arow[(long)(k0 + 64 < K ? k0 + 64 : 0) * sAk], 0, 1);
    v16h a0{}, b0{}, a1{}, b1v{};
    load_full(k0, a0, b0);
    load_full(k0 + 32, a1, b1v);
    acc = __builtin_amdgcn_wmma_f32_16x16x32_f16(false, a0, false, b0,
                                                 (short)0, acc, false, false);
    acc = __builtin_amdgcn_wmma_f32_16x16x32_f16(false, a1, false, b1v,
                                                 (short)0, acc, false, false);
  }
  for (; k0 + 32 <= K; k0 += 32) {
    v16h a{}, b{};
    load_full(k0, a, b);
    acc = __builtin_amdgcn_wmma_f32_16x16x32_f16(false, a, false, b,
                                                 (short)0, acc, false, false);
  }
  if (k0 < K) {
    v16h a{}, b{};
    load_tail(k0, a, b);
    acc = __builtin_amdgcn_wmma_f32_16x16x32_f16(false, a, false, b,
                                                 (short)0, acc, false, false);
  }

  const int dn    = n0 + (lane & 15);
  const int dnc   = dn < N ? dn : (N - 1);
  const int mbase = m0 + (lane >> 4) * 8;
  float bb = 0.f;
  if (bias != nullptr) bb = bias[dnc] * ((dn < N) ? 1.f : 0.f);
#pragma unroll
  for (int v = 0; v < 8; ++v) {
    int dm = mbase + v;
    if (dm < M && dn < N)
      Cb[(long)dm * sCm + (long)dn * sCn] = alpha * acc[v] + bb;
  }
}

// ---------------------------------------------------------------------------
// Implicit-im2col convolution (NCHW, OIHW weights), one image per blockIdx.z.
// M = outH*outW (spatial), N = outC, K = inC*kh*kw, WMMA f16 16x16x32.
// Padding handled with clamped unconditional loads + mask-multiply.
// Also serves conv1d with inW = kw = 1.
// ---------------------------------------------------------------------------
__global__ void wmma_conv2d_kernel(
    const float* __restrict__ in, const float* __restrict__ w,
    float* __restrict__ out,
    int inC, int inH, int inW,
    int outC, int outH, int outW,
    int kh, int kw, int sh, int sw, int ph, int pw)
{
  const int lane = threadIdx.x;
  const int M = outH * outW;
  const int K = inC * kh * kw;
  const int m0 = blockIdx.x * 16;
  const int n0 = (blockIdx.y * blockDim.y + threadIdx.y) * 16;
  if (n0 >= outC) return;

  const int img = blockIdx.z;
  const float* inb  = in  + (long)img * inC * inH * inW;
  float*       outb = out + (long)img * outC * outH * outW;

  const int  am  = m0 + (lane & 15);
  const bool amv = (am < M);
  const int  amc = amv ? am : (M - 1);
  const int  oy  = amc / outW, ox = amc % outW;
  const int  bn  = n0 + (lane & 15);
  const bool bnv = (bn < outC);
  const int  bnc = bnv ? bn : (outC - 1);
  const int  khw = kh * kw;
  const float maskA = amv ? 1.f : 0.f;
  const float maskB = bnv ? 1.f : 0.f;
  const float* wrow = w + (long)bnc * K;

  auto load_ab = [&](int k0, v16h& a, v16h& b) {
#pragma unroll
    for (int e = 0; e < 16; ++e) {
      int ka  = k0 + a_k_of(lane, e);
      int kac = (ka < K) ? ka : (K - 1);
      int c = kac / khw, r = kac % khw;
      int ky = r / kw, kx = r % kw;
      int iy = oy * sh - ph + ky;
      int ix = ox * sw - pw + kx;
      float ma = ((ka < K) && (iy >= 0) && (iy < inH) && (ix >= 0) && (ix < inW))
                     ? maskA : 0.f;
      int iyc = iy < 0 ? 0 : (iy >= inH ? inH - 1 : iy);
      int ixc = ix < 0 ? 0 : (ix >= inW ? inW - 1 : ix);
      float xa = inb[((long)c * inH + iyc) * inW + ixc];
      a[e] = (_Float16)(xa * ma);
      int kb  = k0 + b_k_of(lane, e);
      int kbc = (kb < K) ? kb : (K - 1);
      float xb = wrow[kbc];
      b[e] = (_Float16)(xb * ((kb < K) ? maskB : 0.f));
    }
  };

  v8f acc = {};
  int k0 = 0;
  for (; k0 + 64 <= K; k0 += 64) {
    v16h a0{}, b0{}, a1{}, b1v{};
    load_ab(k0, a0, b0);
    load_ab(k0 + 32, a1, b1v);
    acc = __builtin_amdgcn_wmma_f32_16x16x32_f16(false, a0, false, b0,
                                                 (short)0, acc, false, false);
    acc = __builtin_amdgcn_wmma_f32_16x16x32_f16(false, a1, false, b1v,
                                                 (short)0, acc, false, false);
  }
  for (; k0 < K; k0 += 32) {
    v16h a{}, b{};
    load_ab(k0, a, b);
    acc = __builtin_amdgcn_wmma_f32_16x16x32_f16(false, a, false, b,
                                                 (short)0, acc, false, false);
  }

  const int dn    = n0 + (lane & 15);
  const int mbase = m0 + (lane >> 4) * 8;
#pragma unroll
  for (int v = 0; v < 8; ++v) {
    int dm = mbase + v;
    if (dm < M && dn < outC) {
      int oy2 = dm / outW, ox2 = dm % outW;
      outb[((long)dn * outH + oy2) * outW + ox2] = acc[v];
    }
  }
}

// ---------------------------------------------------------------------------
// Elementwise / reduction kernels
// ---------------------------------------------------------------------------
__global__ void inorm_prelu_kernel(float* __restrict__ x, const float* __restrict__ g,
                                   const float* __restrict__ bt, const float* __restrict__ al,
                                   int C, int S, int do_prelu)
{
  const int row = blockIdx.x;               // over B*C
  const int c = row % C;
  float* p = x + (long)row * S;
  __shared__ float s1[256], s2[256];
  float s = 0.f, q = 0.f;
  for (int i = threadIdx.x; i < S; i += blockDim.x) { float v = p[i]; s += v; q += v * v; }
  s1[threadIdx.x] = s; s2[threadIdx.x] = q;
  __syncthreads();
  for (int o = blockDim.x / 2; o > 0; o >>= 1) {
    if ((int)threadIdx.x < o) { s1[threadIdx.x] += s1[threadIdx.x + o]; s2[threadIdx.x] += s2[threadIdx.x + o]; }
    __syncthreads();
  }
  float mean = s1[0] / S;
  float var  = s2[0] / S - mean * mean;
  float rs   = rsqrtf(var + 1e-5f);
  float gg = g[c], bb = bt[c];
  float a = do_prelu ? al[0] : 0.f;
  for (int i = threadIdx.x; i < S; i += blockDim.x) {
    float v = (p[i] - mean) * rs * gg + bb;
    if (do_prelu && v < 0.f) v *= a;
    p[i] = v;
  }
}

__global__ void layernorm_kernel(float* __restrict__ x, const float* __restrict__ g,
                                 const float* __restrict__ b, int L)
{
  const int row = blockIdx.x;
  float* p = x + (long)row * L;
  __shared__ float s1[256], s2[256];
  float s = 0.f, q = 0.f;
  for (int i = threadIdx.x; i < L; i += blockDim.x) { float v = p[i]; s += v; q += v * v; }
  s1[threadIdx.x] = s; s2[threadIdx.x] = q;
  __syncthreads();
  for (int o = blockDim.x / 2; o > 0; o >>= 1) {
    if ((int)threadIdx.x < o) { s1[threadIdx.x] += s1[threadIdx.x + o]; s2[threadIdx.x] += s2[threadIdx.x + o]; }
    __syncthreads();
  }
  float m = s1[0] / L, var = s2[0] / L - m * m, rs = rsqrtf(var + 1e-5f);
  for (int i = threadIdx.x; i < L; i += blockDim.x) p[i] = (p[i] - m) * rs * g[i] + b[i];
}

__global__ void softmax_rows_kernel(float* __restrict__ x, int L)
{
  const int row = blockIdx.x;
  float* p = x + (long)row * L;
  __shared__ float red[256];
  float mx = -1e30f;
  for (int i = threadIdx.x; i < L; i += blockDim.x) mx = fmaxf(mx, p[i]);
  red[threadIdx.x] = mx; __syncthreads();
  for (int o = blockDim.x / 2; o > 0; o >>= 1) {
    if ((int)threadIdx.x < o) red[threadIdx.x] = fmaxf(red[threadIdx.x], red[threadIdx.x + o]);
    __syncthreads();
  }
  mx = red[0]; __syncthreads();
  float s = 0.f;
  for (int i = threadIdx.x; i < L; i += blockDim.x) { float e = __expf(p[i] - mx); p[i] = e; s += e; }
  red[threadIdx.x] = s; __syncthreads();
  for (int o = blockDim.x / 2; o > 0; o >>= 1) {
    if ((int)threadIdx.x < o) red[threadIdx.x] += red[threadIdx.x + o];
    __syncthreads();
  }
  float inv = 1.f / red[0];
  for (int i = threadIdx.x; i < L; i += blockDim.x) p[i] *= inv;
}

__global__ void sigmoid_kernel(float* x, long n)
{
  long i = (long)blockIdx.x * blockDim.x + threadIdx.x;
  if (i < n) x[i] = 1.f / (1.f + __expf(-x[i]));
}

__global__ void add_inplace_kernel(float* x, const float* y, long n)
{
  long i = (long)blockIdx.x * blockDim.x + threadIdx.x;
  if (i < n) x[i] += y[i];
}

__global__ void fill_kernel(float* x, float v, long n)
{
  long i = (long)blockIdx.x * blockDim.x + threadIdx.x;
  if (i < n) x[i] = v;
}

// xcat[:,0:32] = x11*x21 ; xcat[:,32:64] = x21*x12   (channel concat)
__global__ void concat_mul_kernel(const float* __restrict__ x11, const float* __restrict__ x12,
                                  const float* __restrict__ x21, float* __restrict__ xcat,
                                  int Cc, int S, long nHalf)
{
  long i = (long)blockIdx.x * blockDim.x + threadIdx.x;
  if (i >= nHalf) return;
  long b = i / ((long)Cc * S);
  long r = i - b * (long)Cc * S;
  long o1 = b * (2L * Cc) * S + r;
  xcat[o1]              = x11[i] * x21[i];
  xcat[o1 + (long)Cc*S] = x21[i] * x12[i];
}

// flat(n): out[(b*T+t), f, c] = n[b, c, f, t]
__global__ void flatten_audio_kernel(const float* __restrict__ in, float* __restrict__ out,
                                     int B, int C, int F, int T)
{
  long n = (long)B * C * F * T;
  long i = (long)blockIdx.x * blockDim.x + threadIdx.x;
  if (i >= n) return;
  int t = i % T; long r = i / T;
  int f = r % F; r /= F;
  int c = r % C; int b = r / C;
  out[(((long)b * T + t) * F + f) * C + c] = in[i];
}

// lip [B,1,32,32,T] -> ximg [(t*B+b),1,32,32]
__global__ void copy_lip_kernel(const float* __restrict__ lip, float* __restrict__ out)
{
  long n = 320L * 1024;
  long i = (long)blockIdx.x * blockDim.x + threadIdx.x;
  if (i >= n) return;
  int px = i % 1024; long r = i / 1024;
  int b = r % 64; int t = r / 64;
  out[i] = lip[((long)b * 1024 + px) * 5 + t];
}

// [N,C,H,W] -> [N, H*W, C]
__global__ void nchw_to_nlc_kernel(const float* __restrict__ in, float* __restrict__ out,
                                   int N, int C, int HW)
{
  long n = (long)N * C * HW;
  long i = (long)blockIdx.x * blockDim.x + threadIdx.x;
  if (i >= n) return;
  int hw = i % HW; long r = i / HW;
  int c = r % C; int b = r / C;
  out[((long)b * HW + hw) * C + c] = in[i];
}

// [B, N, C] -> [B, C, N]
__global__ void transpose12_kernel(const float* __restrict__ in, float* __restrict__ out,
                                   int Bt, int N, int C)
{
  long n = (long)Bt * N * C;
  long i = (long)blockIdx.x * blockDim.x + threadIdx.x;
  if (i >= n) return;
  int c = i % C; long r = i / C;
  int nn = r % N; int b = r / N;
  out[((long)b * C + c) * N + nn] = in[i];
}

// attn_av = softmax(attn_qk * lipv, over hd); out = attn_qk * attn_av,
// written directly in [Bt, N, C=H*hd] layout (head h -> cols h*hd..).
__global__ void attn_av_kernel(const float* __restrict__ aq, const float* __restrict__ lipv,
                               float* __restrict__ avout, int Bt, int H, int Ntok, int hd)
{
  long n = (long)Bt * H * Ntok;
  long i = (long)blockIdx.x * blockDim.x + threadIdx.x;
  if (i >= n) return;
  int nn = i % Ntok; long r = i / Ntok;
  int h = r % H; int b = r / H;
  const float* aqp = aq   + (((long)(b * H + h) * Ntok) + nn) * hd;
  const float* lvp = lipv + ((long)b * Ntok + nn) * (H * hd) + h * hd;
  float pv[8]; float mx = -1e30f;
  for (int d = 0; d < hd; ++d) { pv[d] = aqp[d] * lvp[d]; mx = fmaxf(mx, pv[d]); }
  float s = 0.f;
  for (int d = 0; d < hd; ++d) { pv[d] = __expf(pv[d] - mx); s += pv[d]; }
  float inv = 1.f / s;
  float* op = avout + ((long)b * Ntok + nn) * (H * hd) + h * hd;
  for (int d = 0; d < hd; ++d) op[d] = aqp[d] * (pv[d] * inv);
}

__global__ void mean_rows_kernel(const float* __restrict__ in, float* __restrict__ out, int L)
{
  int row = blockIdx.x;
  if (threadIdx.x == 0) {
    float s = 0.f;
    for (int i = 0; i < L; ++i) s += in[(long)row * L + i];
    out[row] = s / L;
  }
}

// out = a * sigmoid(t[bc]) * s + b * x3    (t is [rows,1])
__global__ void gate_fuse_kernel(const float* __restrict__ s, const float* __restrict__ t,
                                 const float* __restrict__ x3, float* __restrict__ out,
                                 const float* pa, const float* pb, int L, long n)
{
  long i = (long)blockIdx.x * blockDim.x + threadIdx.x;
  if (i >= n) return;
  long bc = i / L;
  float sg = 1.f / (1.f + __expf(-t[bc]));
  out[i] = pa[0] * (sg * s[i]) + pb[0] * x3[i];
}

__global__ void lstm_cell_kernel(const float* __restrict__ g1, const float* __restrict__ g2,
                                 float* __restrict__ h, float* __restrict__ c,
                                 float* __restrict__ out, int B, int Hd, int T, int t,
                                 int dir_off, int outC)
{
  int i = blockIdx.x * blockDim.x + threadIdx.x;
  if (i >= B * Hd) return;
  int b = i / Hd, j = i % Hd;
  const float* a1 = g1 + (long)b * 4 * Hd;
  const float* a2 = g2 + (long)b * 4 * Hd;
  float ig = a1[j]          + a2[j];
  float fg = a1[Hd + j]     + a2[Hd + j];
  float gv = a1[2 * Hd + j] + a2[2 * Hd + j];
  float og = a1[3 * Hd + j] + a2[3 * Hd + j];
  float si = 1.f / (1.f + __expf(-ig));
  float sf = 1.f / (1.f + __expf(-fg));
  float so = 1.f / (1.f + __expf(-og));
  float cc = sf * c[i] + si * tanhf(gv);
  float hh = so * tanhf(cc);
  c[i] = cc; h[i] = hh;
  out[((long)b * T + t) * outC + dir_off + j] = hh;
}

// ---------------------------------------------------------------------------
// Host orchestration
// ---------------------------------------------------------------------------
extern "C" void kernel_launch(void* const* d_in, const int* in_sizes, int n_in,
                              void* d_out, int out_size, void* d_ws, size_t ws_size,
                              hipStream_t stream)
{
  (void)in_sizes; (void)n_in; (void)out_size; (void)ws_size;
  const float* P[80];
  for (int i = 0; i < 80 && i < n_in; ++i) P[i] = (const float*)d_in[i];
  const float* noisy = P[0];
  const float* lip   = P[1];
  // pytree (sorted-key) flat order of params:
  //  2: a_fc3_w ; 3..13 ac1 {in_a,in_b,in_g,in_w,out_a,out_b,out_g,out_w,w11,w12,w21}
  //  14..24 ac2 ; 25..35 ac3 ; 36..40 att {lipv_w,proj_b,proj_w,qkv_w,se_w} ; 41 fc2_w
  //  42..56 gc {a,a11,a12,a2,b,b11,b12,b2,g11,g12,g2,w11,w12,w2,w31}
  //  57..72 img {a2,a4,a6,b2,b4,b6,g2,g4,g6,w11,w12,w2,w3,w4,w5,w6}
  //  73 ln_b ; 74 ln_g ; 75..78 lstm {hh_b,hh_f,ih_b,ih_f} ; 79 v_fc3_w

  size_t off = 0;
  auto alloc = [&](long elems) -> float* {
    float* p = (float*)((char*)d_ws + off);
    off += (((size_t)elems * sizeof(float) + 255) & ~(size_t)255);
    return p;
  };

  // ----- workspace -----
  float* xbuf   = alloc(64L * 32 * 256 * 5);
  float* x11b   = alloc(64L * 32 * 256 * 5);
  float* x12b   = alloc(64L * 32 * 256 * 5);
  float* x21b   = alloc(64L * 32 * 256 * 5);
  float* xcatb  = alloc(64L * 64 * 256 * 5);
  float* n1b    = alloc(64L * 32 * 128 * 5);
  float* n2b    = alloc(64L * 32 * 64 * 5);
  float* n3b    = alloc(64L * 32 * 32 * 5);
  float* aud1   = alloc(320L * 128 * 32);
  float* aud2   = alloc(320L * 64 * 32);
  float* aud3   = alloc(320L * 32 * 32);
  float* ximg   = alloc(320L * 1024);
  float* it1    = alloc(320L * 32 * 16 * 32);
  float* it2    = alloc(320L * 32 * 16 * 16);
  float* l1buf  = alloc(320L * 32 * 8 * 16);
  float* l2buf  = alloc(320L * 32 * 8 * 8);
  float* l3buf  = alloc(320L * 32 * 4 * 8);
  float* lm1    = alloc(320L * 128 * 32);
  float* lm2    = alloc(320L * 64 * 32);
  float* lm3    = alloc(320L * 32 * 32);
  float* qkvb   = alloc(320L * 128 * 96);
  float* scob   = alloc(320L * 4 * 128 * 128);
  float* aqb    = alloc(320L * 4 * 128 * 8);
  float* lipvb  = alloc(320L * 128 * 32);
  float* avtb   = alloc(320L * 128 * 32);
  float* avpb   = alloc(320L * 128 * 32);
  float* qlb    = alloc(320L * 128 * 32);
  float* seb    = alloc(320L * 32 * 32);
  float* outmb  = alloc(320L * 128 * 32);
  float* av1t   = alloc(320L * 32 * 128);
  float* av2t   = alloc(320L * 32 * 64);
  float* av3t   = alloc(320L * 32 * 32);
  float* x1ab   = alloc(320L * 64 * 64);
  float* x1bb   = alloc(320L * 32 * 32);
  float* x2ab   = alloc(320L * 32 * 32);
  float* ttb    = alloc(320L * 32 * 16);
  float* tmb    = alloc(320L * 32);
  float* tqb    = alloc(320L * 32);
  float* avgb   = alloc(320L * 32 * 32);
  float* g1b    = alloc(64L * 512);
  float* g2b    = alloc(64L * 512);
  float* hb     = alloc(64L * 128);
  float* cbuf   = alloc(64L * 128);
  float* lstmo  = alloc(64L * 5 * 256);
  float* fcb    = alloc(64L * 512);

  auto egrid = [](long n) { return dim3((unsigned)((n + 255) / 256)); };

  auto gemm = [&](const float* A, const float* Bm, const float* bias, float* C,
                  int M, int N, int K,
                  int sAm, int sAk, int sBk, int sBn, int sCm, int sCn,
                  int batch, long bsA, long bsB, long bsC,
                  int inner, long bsA2, long bsB2, long bsC2, float alpha) {
    dim3 grid((M + 15) / 16, (N + 63) / 64, batch);
    wmma_gemm_kernel<<<grid, dim3(32, 4, 1), 0, stream>>>(
        A, Bm, bias, C, M, N, K, sAm, sAk, sBk, sBn, sCm, sCn,
        bsA, bsB, bsC, inner, bsA2, bsB2, bsC2, alpha);
  };

  auto conv2d = [&](const float* in, const float* w, float* out, int Bn,
                    int inC, int inH, int inW, int outC,
                    int kh, int kw, int sh, int sw, int ph, int pw) {
    int outH = (inH + 2 * ph - kh) / sh + 1;
    int outW = (inW + 2 * pw - kw) / sw + 1;
    int M = outH * outW;
    dim3 grid((M + 15) / 16, (outC + 63) / 64, Bn);
    wmma_conv2d_kernel<<<grid, dim3(32, 4, 1), 0, stream>>>(
        in, w, out, inC, inH, inW, outC, outH, outW, kh, kw, sh, sw, ph, pw);
  };
  auto conv1d = [&](const float* in, const float* w, float* out, int Bn,
                    int inC, int L, int outC, int k, int s, int p) {
    conv2d(in, w, out, Bn, inC, L, 1, outC, k, 1, s, 1, p, 0);
  };

  auto inorm = [&](float* x, const float* g, const float* b, const float* a,
                   int rows, int C, int S) {
    inorm_prelu_kernel<<<dim3(rows), dim3(256), 0, stream>>>(x, g, b, a, C, S, 1);
  };
  auto sigm = [&](float* x, long n) {
    sigmoid_kernel<<<egrid(n), dim3(256), 0, stream>>>(x, n);
  };

  // ---------------- engate_audio ----------------
  auto engate = [&](const float* xin, int inC, int H, int l, int pb, float* nout) {
    conv2d(xin, P[pb + 3], xbuf, 64, inC, H, 5, 32, 1, 1, 1, 1, 0, 0);          // in_w
    inorm(xbuf, P[pb + 2], P[pb + 1], P[pb + 0], 64 * 32, 32, H * 5);
    conv2d(xbuf, P[pb + 8], x11b, 64, 32, H, 5, 32, l, 1, 1, 1, l / 2, 0);      // w11
    conv2d(x11b, P[pb + 9], x12b, 64, 32, H, 5, 32, 1, 5, 1, 1, 0, 2);          // w12
    sigm(x12b, 64L * 32 * H * 5);
    conv2d(xbuf, P[pb + 10], x21b, 64, 32, H, 5, 32, l / 2 + 1, 3, 1, 1, l / 4, 1); // w21
    concat_mul_kernel<<<egrid(64L * 32 * H * 5), dim3(256), 0, stream>>>(
        x11b, x12b, x21b, xcatb, 32, H * 5, 64L * 32 * H * 5);
    conv2d(xcatb, P[pb + 7], nout, 64, 64, H, 5, 32, 1, 1, 2, 1, 0, 0);         // out_w
    inorm(nout, P[pb + 6], P[pb + 5], P[pb + 4], 64 * 32, 32, (H / 2) * 5);
  };
  engate(noisy, 1, 256, 13, 3, n1b);
  engate(n1b,  32, 128,  9, 14, n2b);
  engate(n2b,  32,  64,  5, 25, n3b);

  flatten_audio_kernel<<<egrid(64L*32*128*5), dim3(256), 0, stream>>>(n1b, aud1, 64, 32, 128, 5);
  flatten_audio_kernel<<<egrid(64L*32*64*5),  dim3(256), 0, stream>>>(n2b, aud2, 64, 32, 64, 5);
  flatten_audio_kernel<<<egrid(64L*32*32*5),  dim3(256), 0, stream>>>(n3b, aud3, 64, 32, 32, 5);

  // ---------------- image tower ----------------
  copy_lip_kernel<<<egrid(320L * 1024), dim3(256), 0, stream>>>(lip, ximg);
  conv2d(ximg, P[66], it1, 320, 1, 32, 32, 32, 5, 5, 2, 1, 2, 2);   // -> [320,32,16,32]
  conv2d(it1,  P[67], it2, 320, 32, 16, 32, 32, 5, 5, 1, 2, 2, 2);  // -> [320,32,16,16]
  conv2d(it2,  P[68], l1buf, 320, 32, 16, 16, 32, 5, 5, 2, 1, 2, 2);// -> [320,32,8,16]
  inorm(l1buf, P[63], P[60], P[57], 320 * 32, 32, 128);
  conv2d(l1buf, P[69], it2, 320, 32, 8, 16, 32, 5, 5, 1, 2, 2, 2);  // -> [320,32,8,8]
  conv2d(it2,   P[70], l2buf, 320, 32, 8, 8, 32, 5, 5, 1, 1, 2, 2); // -> [320,32,8,8]
  inorm(l2buf, P[64], P[61], P[58], 320 * 32, 32, 64);
  conv2d(l2buf, P[71], it2, 320, 32, 8, 8, 32, 3, 3, 2, 1, 1, 1);   // -> [320,32,4,8]
  conv2d(it2,   P[72], l3buf, 320, 32, 4, 8, 32, 3, 3, 1, 1, 1, 1); // -> [320,32,4,8]
  inorm(l3buf, P[65], P[62], P[59], 320 * 32, 32, 32);
  nchw_to_nlc_kernel<<<egrid(320L*32*128), dim3(256), 0, stream>>>(l1buf, lm1, 320, 32, 128);
  nchw_to_nlc_kernel<<<egrid(320L*32*64),  dim3(256), 0, stream>>>(l2buf, lm2, 320, 32, 64);
  nchw_to_nlc_kernel<<<egrid(320L*32*32),  dim3(256), 0, stream>>>(l3buf, lm3, 320, 32, 32);

  // ---------------- av attention ----------------
  auto attention = [&](const float* audia, const float* lmat, int Nt, float* avout_t) {
    const int BT = 320, Hh = 4, hd = 8;
    const long rows = (long)BT * Nt;
    const long NN = (long)Nt * Nt;
    // qkv = audia @ qkv_w.T    [rows,96]
    gemm(audia, P[39], nullptr, qkvb, (int)rows, 96, 32,
         32, 1, 1, 32, 96, 1, 1, 0, 0, 0, 1, 0, 0, 0, 1.f);
    // scores[b,h] = q @ k.T * hd^-0.5
    gemm(qkvb, qkvb + 32, nullptr, scob, Nt, Nt, hd,
         96, 1, 1, 96, Nt, 1,
         BT * Hh, (long)Nt * 96, (long)Nt * 96, Hh * NN,
         Hh, 8, 8, NN, 0.35355339059327f);
    softmax_rows_kernel<<<dim3((unsigned)(BT * Hh * Nt)), dim3(256), 0, stream>>>(scob, Nt);
    // attn_qk = attn @ k       [b,h,Nt,hd]
    gemm(scob, qkvb + 32, nullptr, aqb, Nt, hd, Nt,
         Nt, 1, 96, 1, hd, 1,
         BT * Hh, Hh * NN, (long)Nt * 96, (long)Hh * Nt * hd,
         Hh, NN, 8, (long)Nt * hd, 1.f);
    // lipv = lip @ lipv_w.T
    gemm(lmat, P[36], nullptr, lipvb, (int)rows, 32, 32,
         32, 1, 1, 32, 32, 1, 1, 0, 0, 0, 1, 0, 0, 0, 1.f);
    attn_av_kernel<<<egrid((long)BT * Hh * Nt), dim3(256), 0, stream>>>(
        aqb, lipvb, avtb, BT, Hh, Nt, hd);
    // av_out = avt @ proj_w.T + proj_b
    gemm(avtb, P[38], P[37], avpb, (int)rows, 32, 32,
         32, 1, 1, 32, 32, 1, 1, 0, 0, 0, 1, 0, 0, 0, 1.f);
    // ql = audia @ se_w.T ; se = sigmoid((2/sqrt(32)) ql^T ql)
    gemm(audia, P[40], nullptr, qlb, (int)rows, 32, 32,
         32, 1, 1, 32, 32, 1, 1, 0, 0, 0, 1, 0, 0, 0, 1.f);
    gemm(qlb, qlb, nullptr, seb, 32, 32, Nt,
         1, 32, 32, 1, 32, 1,
         BT, (long)Nt * 32, (long)Nt * 32, 1024,
         1, 0, 0, 0, 2.f / sqrtf(32.f));
    sigm(seb, 320L * 1024);
    // out = av_out @ se (per batch)
    gemm(avpb, seb, nullptr, outmb, Nt, 32, 32,
         32, 1, 32, 1, 32, 1,
         BT, (long)Nt * 32, 1024, (long)Nt * 32,
         1, 0, 0, 0, 1.f);
    transpose12_kernel<<<egrid((long)BT * Nt * 32), dim3(256), 0, stream>>>(
        outmb, avout_t, BT, Nt, 32);
  };
  attention(aud1, lm1, 128, av1t);
  attention(aud2, lm2,  64, av2t);
  attention(aud3, lm3,  32, av3t);

  // ---------------- gate conv ----------------
  conv1d(av1t, P[53], x1ab, 320, 32, 128, 64, 13, 2, 6);             // -> [320,64,64]
  inorm(x1ab, P[50], P[47], P[43], 320 * 64, 64, 64);
  conv1d(x1ab, P[54], x1bb, 320, 64, 64, 32, 13, 2, 6);              // -> [320,32,32]
  inorm(x1bb, P[51], P[48], P[44], 320 * 32, 32, 32);
  conv1d(av2t, P[55], x2ab, 320, 32, 64, 32, 9, 2, 4);               // -> [320,32,32]
  inorm(x2ab, P[52], P[49], P[45], 320 * 32, 32, 32);
  add_inplace_kernel<<<egrid(320L*32*32), dim3(256), 0, stream>>>(x1bb, x2ab, 320L * 32 * 32);
  conv1d(x1bb, P[56], ttb, 320, 32, 32, 32, 9, 2, 4);                // -> [320,32,16]
  mean_rows_kernel<<<dim3(320 * 32), dim3(32), 0, stream>>>(ttb, tmb, 16);
  conv1d(tmb, P[56], tqb, 320, 32, 1, 32, 9, 2, 4);                  // -> [320,32,1]
  gate_fuse_kernel<<<egrid(320L*32*32), dim3(256), 0, stream>>>(
      x1bb, tqb, av3t, avgb, P[42], P[46], 32, 320L * 32 * 32);

  // ---------------- BiLSTM ----------------
  fill_kernel<<<egrid(64L*128), dim3(256), 0, stream>>>(hb, 0.f, 64L * 128);
  fill_kernel<<<egrid(64L*128), dim3(256), 0, stream>>>(cbuf, 0.f, 64L * 128);
  for (int t = 0; t < 5; ++t) {
    gemm(avgb + (long)t * 1024, P[78], nullptr, g1b, 64, 512, 1024,
         5 * 1024, 1, 1, 1024, 512, 1, 1, 0, 0, 0, 1, 0, 0, 0, 1.f);   // ih_f
    gemm(hb, P[76], nullptr, g2b, 64, 512, 128,
         128, 1, 1, 128, 512, 1, 1, 0, 0, 0, 1, 0, 0, 0, 1.f);          // hh_f
    lstm_cell_kernel<<<egrid(64L*128), dim3(256), 0, stream>>>(
        g1b, g2b, hb, cbuf, lstmo, 64, 128, 5, t, 0, 256);
  }
  fill_kernel<<<egrid(64L*128), dim3(256), 0, stream>>>(hb, 0.f, 64L * 128);
  fill_kernel<<<egrid(64L*128), dim3(256), 0, stream>>>(cbuf, 0.f, 64L * 128);
  for (int t = 4; t >= 0; --t) {
    gemm(avgb + (long)t * 1024, P[77], nullptr, g1b, 64, 512, 1024,
         5 * 1024, 1, 1, 1024, 512, 1, 1, 0, 0, 0, 1, 0, 0, 0, 1.f);   // ih_b
    gemm(hb, P[75], nullptr, g2b, 64, 512, 128,
         128, 1, 1, 128, 512, 1, 1, 0, 0, 0, 1, 0, 0, 0, 1.f);          // hh_b
    lstm_cell_kernel<<<egrid(64L*128), dim3(256), 0, stream>>>(
        g1b, g2b, hb, cbuf, lstmo, 64, 128, 5, t, 128, 256);
  }

  // ---------------- head ----------------
  gemm(lstmo, P[41], nullptr, fcb, 64, 512, 1280,
       1280, 1, 1, 1280, 512, 1, 1, 0, 0, 0, 1, 0, 0, 0, 1.f);          // fc2
  layernorm_kernel<<<dim3(64), dim3(256), 0, stream>>>(fcb, P[74], P[73], 512);
  float* out = (float*)d_out;
  gemm(fcb, P[2], nullptr, out, 64, 256, 512,
       512, 1, 1, 512, 256, 1, 1, 0, 0, 0, 1, 0, 0, 0, 1.f);            // a_fc3
  gemm(fcb, P[79], nullptr, out + 64L * 256, 64, 2048, 512,
       512, 1, 1, 512, 2048, 1, 1, 0, 0, 0, 1, 0, 0, 0, 1.f);           // v_fc3
}
